// GeometricFeatureProcessor_49435073577252
// MI455X (gfx1250) — matrix operand order repaired and checked by reading, not compiled, not run on hardware
//
#include <hip/hip_runtime.h>

// ---------------------------------------------------------------------------
// GeometricFeatureProcessor for MI455X (gfx1250, wave32, WMMA)
//   B=2, N=8192, K=32 neighbors, C=16 feats, MLP 512->256->128->256
// ---------------------------------------------------------------------------

typedef _Float16 v16h __attribute__((ext_vector_type(16)));
typedef _Float16 v8h  __attribute__((ext_vector_type(8)));
typedef float    v8f  __attribute__((ext_vector_type(8)));

#define NPTS     8192
#define CFEAT    16
#define KNEI     32
#define INFDIM   512
#define H1DIM    256
#define H2DIM    128
#define OUTDIM   256
#define RADIUS2  0.01f
#define CAP      128           // candidate capacity per query (E[count]~34)

#define XS_STRIDE  520         // 512 + 8 halves pad (keeps 16B align, spreads banks)
#define H1_STRIDE  264
#define H2_STRIDE  136

// Branch-free tanh: 1 - 2/(exp(2x)+1).  v_exp_f32 saturates to +inf / 0 for
// large |x|, so the formula converges to +/-1 with no divergent slow path.
__device__ __forceinline__ float fast_tanh(float x) {
    float e = __expf(2.0f * x);                       // 1x v_exp_f32
    float r = __builtin_amdgcn_rcpf(e + 1.0f);        // 1x v_rcp_f32
    return __builtin_fmaf(-2.0f, r, 1.0f);
}

__device__ __forceinline__ float gelu_tanh(float x) {
    const float c0 = 0.7978845608028654f;   // sqrt(2/pi)
    float x3 = x * x * x;
    return 0.5f * x * (1.0f + fast_tanh(c0 * (x + 0.044715f * x3)));
}

// ---------------------------------------------------------------------------
// Prep: pack W1/W2/W3 (f32 row-major [K][N]) into f16 column-major [N][K]
// so each lane's WMMA B-fragment is one contiguous 32B load.
// ---------------------------------------------------------------------------
__global__ void pack_weights_kernel(const float* __restrict__ W1,
                                    const float* __restrict__ W2,
                                    const float* __restrict__ W3,
                                    _Float16* __restrict__ W1h,
                                    _Float16* __restrict__ W2h,
                                    _Float16* __restrict__ W3h) {
    int i = blockIdx.x * blockDim.x + threadIdx.x;
    const int S1 = INFDIM * H1DIM;          // 131072
    const int S2 = H1DIM * H2DIM;           // 32768
    const int S3 = H2DIM * OUTDIM;          // 32768
    if (i < S1) {
        int n = i / INFDIM, k = i % INFDIM;
        W1h[i] = (_Float16)W1[k * H1DIM + n];
    } else if (i < S1 + S2) {
        int j = i - S1;
        int n = j / H1DIM, k = j % H1DIM;
        W2h[j] = (_Float16)W2[k * H2DIM + n];
    } else if (i < S1 + S2 + S3) {
        int j = i - S1 - S2;
        int n = j / H2DIM, k = j % H2DIM;
        W3h[j] = (_Float16)W3[k * OUTDIM + n];
    }
}

// ---------------------------------------------------------------------------
// A-fragment builder: 16-bit A 16x32, contiguous k order used here (not the ISA table order; B matches):
//   lanes 0-15 : M=lane,    halves 0-15 = K kb+0..15
//   lanes 16-31: M=lane-16, halves 0-15 = K kb+16..31
// src = &row_base[kb], 16B-aligned, koff = (lane&16)?16:0
// ---------------------------------------------------------------------------
__device__ __forceinline__ v16h load_a_frag(const _Float16* src, int koff) {
    v8h lo = *(const v8h*)(src + koff);
    v8h hi = *(const v8h*)(src + 8 + koff);
    v16h a;
#pragma unroll
    for (int i = 0; i < 8; ++i) { a[i] = lo[i]; a[8 + i] = hi[i]; }
    return a;
}

// ---------------------------------------------------------------------------
// Fused ball-query + gather + 3-layer MLP. One block = 16 query points.
// ---------------------------------------------------------------------------
__global__ __launch_bounds__(256)
void geo_mlp_fused_kernel(const float* __restrict__ pts,     // (B,N,3)
                          const float* __restrict__ feats,   // (B,N,16)
                          const _Float16* __restrict__ W1h,  // [256][512] col-major f16
                          const float* __restrict__ b1,
                          const _Float16* __restrict__ W2h,  // [128][256]
                          const float* __restrict__ b2,
                          const _Float16* __restrict__ W3h,  // [256][128]
                          const float* __restrict__ b3,
                          float* __restrict__ out) {         // (B,N,256)
    __shared__ _Float16 Xs [16][XS_STRIDE];
    __shared__ _Float16 H1s[16][H1_STRIDE];
    __shared__ _Float16 H2s[16][H2_STRIDE];
    __shared__ float    cd2 [16][CAP];
    __shared__ int      cidx[16][CAP];

    const int lane = threadIdx.x & 31;
    const int wave = threadIdx.x >> 5;
    const int p0   = blockIdx.x * 16;          // global point tile (same batch: 8192%16==0)
    const int b    = p0 / NPTS;
    const int n0   = p0 % NPTS;
    const float* ptsb = pts   + (size_t)b * NPTS * 3;
    const float* ftb  = feats + (size_t)b * NPTS * CFEAT;

    // ---- zero the X tile (zero slots == masked / absent neighbors) ----
    for (int i = threadIdx.x; i < 16 * XS_STRIDE; i += 256)
        (&Xs[0][0])[i] = (_Float16)0.f;
    __syncthreads();

    // =========================== Phase A: ball query ========================
    // Each wave handles 2 query points. Out-of-radius top-k entries are
    // masked to zero in the reference, and in-radius points always outrank
    // them, so: collect in-radius candidates, sort by (d2,idx), keep first 32.
    for (int qq = 0; qq < 2; ++qq) {
        const int q  = wave * 2 + qq;          // local row 0..15
        const int nq = n0 + q;
        const float qx = ptsb[nq * 3 + 0];
        const float qy = ptsb[nq * 3 + 1];
        const float qz = ptsb[nq * 3 + 2];

        int base = 0;                          // uniform across the wave
        for (int it = 0; it < NPTS / 32; ++it) {
            const int j = it * 32 + lane;
            float dx = ptsb[j * 3 + 0] - qx;
            float dy = ptsb[j * 3 + 1] - qy;
            float dz = ptsb[j * 3 + 2] - qz;
            float d2 = dx * dx + dy * dy + dz * dz;
            bool hit = (d2 <= RADIUS2);
            unsigned mask = (unsigned)__ballot(hit);
            if (hit) {
                int pos = base + __popc(mask & ((1u << lane) - 1u));
                if (pos < CAP) { cd2[q][pos] = d2; cidx[q][pos] = j; }
            }
            base += __popc(mask);
        }
        const int M = (base < CAP) ? base : CAP;

        __syncthreads();   // uniform: orders cand writes vs reads below

        // rank-by-count (M ~ 34) and gather features into slot = rank
        for (int c = lane; c < M; c += 32) {
            float dc = cd2[q][c];
            int   ic = cidx[q][c];
            int rank = 0;
            for (int j2 = 0; j2 < M; ++j2) {
                float dj = cd2[q][j2];
                int   ij = cidx[q][j2];
                rank += (dj < dc) || (dj == dc && ij < ic);
            }
            if (rank < KNEI) {
                const float* f = ftb + (size_t)ic * CFEAT;
#pragma unroll
                for (int cc = 0; cc < CFEAT; ++cc)
                    Xs[q][rank * CFEAT + cc] = (_Float16)f[cc];
            }
        }
    }
    __syncthreads();

    const int row  = lane & 15;                // A row; B/D column within tile
    const int koff = (lane & 16) ? 16 : 0;     // A-fragment K sub-offset
    const int boff = (lane & 16) ? 16 : 0;     // B-fragment K sub-offset
    const int mofs = (lane & 16) ? 8 : 0;      // C/D row offset

    // ================= Layer 1: X(16x512) @ W1 -> gelu -> H1 ================
    {
#pragma unroll 1
        for (int t = 0; t < 2; ++t) {          // 8 waves x 2 = 16 col-tiles
            const int col = (wave * 2 + t) * 16 + row;
            v8f acc = {};
#pragma unroll 4
            for (int kb = 0; kb < INFDIM; kb += 32) {
                v16h a  = load_a_frag(&Xs[row][kb], koff);
                v16h bf = *(const v16h*)(W1h + (size_t)col * INFDIM + kb + boff);
                acc = __builtin_amdgcn_wmma_f32_16x16x32_f16(
                          false, a, false, bf, (short)0, acc, false, false);
            }
            const float bias = b1[col];
#pragma unroll
            for (int j = 0; j < 8; ++j)
                H1s[mofs + j][col] = (_Float16)gelu_tanh(acc[j] + bias);
        }
    }
    __syncthreads();

    // ================ Layer 2: H1(16x256) @ W2 -> gelu -> H2 ================
    {
        const int col = wave * 16 + row;       // 8 waves x 1 = 8 col-tiles
        v8f acc = {};
#pragma unroll 4
        for (int kb = 0; kb < H1DIM; kb += 32) {
            v16h a  = load_a_frag(&H1s[row][kb], koff);
            v16h bf = *(const v16h*)(W2h + (size_t)col * H1DIM + kb + boff);
            acc = __builtin_amdgcn_wmma_f32_16x16x32_f16(
                      false, a, false, bf, (short)0, acc, false, false);
        }
        const float bias = b2[col];
#pragma unroll
        for (int j = 0; j < 8; ++j)
            H2s[mofs + j][col] = (_Float16)gelu_tanh(acc[j] + bias);
    }
    __syncthreads();

    // ================ Layer 3: H2(16x128) @ W3 -> tanh -> out ===============
    {
#pragma unroll 1
        for (int t = 0; t < 2; ++t) {          // 8 waves x 2 = 16 col-tiles
            const int col = (wave * 2 + t) * 16 + row;
            v8f acc = {};
#pragma unroll
            for (int kb = 0; kb < H2DIM; kb += 32) {
                v16h a  = load_a_frag(&H2s[row][kb], koff);
                v16h bf = *(const v16h*)(W3h + (size_t)col * H2DIM + kb + boff);
                acc = __builtin_amdgcn_wmma_f32_16x16x32_f16(
                          false, a, false, bf, (short)0, acc, false, false);
            }
            const float bias = b3[col];
#pragma unroll
            for (int j = 0; j < 8; ++j)
                out[(size_t)(p0 + mofs + j) * OUTDIM + col] =
                    fast_tanh(acc[j] + bias);
        }
    }
}

// ---------------------------------------------------------------------------
extern "C" void kernel_launch(void* const* d_in, const int* in_sizes, int n_in,
                              void* d_out, int out_size, void* d_ws, size_t ws_size,
                              hipStream_t stream) {
    (void)in_sizes; (void)n_in; (void)out_size; (void)ws_size;
    const float* pts   = (const float*)d_in[0];   // (2,8192,3)
    const float* feats = (const float*)d_in[1];   // (2,8192,16)
    const float* W1    = (const float*)d_in[2];   // (512,256)
    const float* b1    = (const float*)d_in[3];   // (256,)
    const float* W2    = (const float*)d_in[4];   // (256,128)
    const float* b2    = (const float*)d_in[5];   // (128,)
    const float* W3    = (const float*)d_in[6];   // (128,256)
    const float* b3    = (const float*)d_in[7];   // (256,)
    float* out = (float*)d_out;                   // (2,8192,256)

    _Float16* W1h = (_Float16*)d_ws;              // 256 KB
    _Float16* W2h = W1h + INFDIM * H1DIM;         //  64 KB
    _Float16* W3h = W2h + H1DIM * H2DIM;          //  64 KB

    const int total = INFDIM * H1DIM + H1DIM * H2DIM + H2DIM * OUTDIM;
    pack_weights_kernel<<<(total + 255) / 256, 256, 0, stream>>>(
        W1, W2, W3, W1h, W2h, W3h);

    const int nblocks = (2 * NPTS) / 16;          // 1024 blocks, 256 thr (8 waves)
    geo_mlp_fused_kernel<<<nblocks, 256, 0, stream>>>(
        pts, feats, W1h, b1, W2h, b2, W3h, b3, out);
}
